// MultiHeadAttention_65395172049448
// MI455X (gfx1250) — matrix operand order repaired and checked by reading, not compiled, not run on hardware
//
#include <hip/hip_runtime.h>
#include <hip/hip_bf16.h>

typedef __bf16 bf16;
typedef __attribute__((ext_vector_type(8)))  __bf16 v8bf;
typedef __attribute__((ext_vector_type(16))) __bf16 v16bf;
typedef __attribute__((ext_vector_type(8)))  float  v8f;
typedef __attribute__((ext_vector_type(4)))  unsigned int u32x4;
typedef __attribute__((ext_vector_type(8)))  int i32x8;
typedef __attribute__((ext_vector_type(4)))  int i32x4;

#define B_   2
#define S_   2048
#define D_   1024
#define H_   16
#define DK_  64
#define BH_  (B_*H_)
#define MROWS (B_*S_)   // 4096

#if __has_builtin(__builtin_amdgcn_tensor_load_to_lds)
#define HAVE_TDM 1
#endif
#if __has_include(<hip/amd_detail/amd_gfx1250_TDM.h>)
#define TDM_6ARG 1
#endif

__device__ __forceinline__ v16bf cat16(v8bf lo, v8bf hi){
  return __builtin_shufflevector(lo, hi, 0,1,2,3,4,5,6,7,8,9,10,11,12,13,14,15);
}
// A-fragment (16x32 bf16): per lane, row m=lane%16, 16 contiguous K at (lane/16)*16
__device__ __forceinline__ v16bf load_frag_a(const bf16* p){
  return cat16(*(const v8bf*)p, *(const v8bf*)(p + 8));
}
// B-fragment (32x16 bf16): per lane, col n=lane%16, 16 contiguous K at (lane/16)*16
__device__ __forceinline__ v16bf load_frag_b(const bf16* p){
  return cat16(*(const v8bf*)p, *(const v8bf*)(p + 8));
}
__device__ __forceinline__ v8f wmma_bf(v16bf a, v16bf b, v8f c){
  return __builtin_amdgcn_wmma_f32_16x16x32_bf16(false, a, false, b, (short)0, c, false, false);
}

__device__ __forceinline__ void wait_tensorcnt0(){
#if __has_builtin(__builtin_amdgcn_s_wait_tensorcnt)
  __builtin_amdgcn_s_wait_tensorcnt(0);
#else
  asm volatile("s_wait_tensorcnt 0x0" ::: "memory");
#endif
}

// ---- Tensor Data Mover: 2D tile (tile_d0 elems/row x tile_d1 rows) global -> LDS ----
// D# packing per CDNA5 ISA. data_size = 2 bytes (bf16).
__device__ __forceinline__ void tdm_load_2d(unsigned lds_off, const void* gptr,
                                            unsigned tensor_d0, unsigned tensor_d1,
                                            unsigned tile_d0, unsigned tile_d1,
                                            unsigned long long stride0_elems){
#if defined(HAVE_TDM)
  unsigned long long ga = (unsigned long long)(uintptr_t)gptr;
  u32x4 g0 = { 1u,                                   // count=1, user mode
               lds_off,                               // lds_addr (bytes)
               (unsigned)(ga & 0xffffffffu),          // global_addr[31:0]
               (unsigned)((ga >> 32) & 0x01ffffffu) | (2u << 30) }; // addr[56:32] | type=2
  i32x8 g1 = { (int)(1u << 16),                                         // data_size=1 (2B)
               (int)((tensor_d0 & 0xffffu) << 16),                      // tensor_dim0 lo16
               (int)((tensor_d0 >> 16) | ((tensor_d1 & 0xffffu) << 16)),// d0 hi16 | d1 lo16
               (int)((tensor_d1 >> 16) | (tile_d0 << 16)),              // d1 hi16 | tile_dim0
               (int)(tile_d1),                                          // tile_dim1 (tile_dim2=0)
               (int)(stride0_elems & 0xffffffffu),                      // dim0_stride[31:0]
               (int)((stride0_elems >> 32) & 0xffffu),                  // dim0_stride[47:32]
               0 };
  i32x4 z4 = {0,0,0,0};
#if defined(TDM_6ARG)
  i32x8 z8 = {0,0,0,0,0,0,0,0};
  __builtin_amdgcn_tensor_load_to_lds(g0, g1, z4, z4, z8, 0);
#else
  __builtin_amdgcn_tensor_load_to_lds(g0, g1, z4, z4, 0);
#endif
#else
  (void)lds_off; (void)gptr; (void)tensor_d0; (void)tensor_d1;
  (void)tile_d0; (void)tile_d1; (void)stride0_elems;
#endif
}

// ---------------- fp32 -> bf16 conversion ----------------
__global__ void cvt_bf16(const float* __restrict__ src, bf16* __restrict__ dst, int n){
  int i = blockIdx.x * blockDim.x + threadIdx.x;
  if (i < n) dst[i] = (bf16)src[i];
}

// ================= GEMM core: 128x128 block tile, TDM double-buffered LDS =================
// X:[Mtot,1024] bf16, W:[1024,1024] bf16 (row-major N,K). 8 waves, each 32x64 sub-tile.
#define KSTEP 32
#define TILE_E (128*KSTEP)    // elements per LDS tile buffer

__device__ __forceinline__ void gemm_tiles(const bf16* __restrict__ X,
                                           const bf16* __restrict__ W,
                                           bf16* Xs, bf16* Ws,          // [2][TILE_E]
                                           int gm0, int gn0, int mw, int nw,
                                           int lm, int lh, int wid,
                                           v8f (&c)[2][4], int mtot){
#if defined(HAVE_TDM)
  if (wid == 0){
    tdm_load_2d((unsigned)(uintptr_t)(void*)Xs, X + (size_t)gm0*D_, D_, (unsigned)mtot, KSTEP, 128, D_);
    tdm_load_2d((unsigned)(uintptr_t)(void*)Ws, W + (size_t)gn0*D_, D_, D_,             KSTEP, 128, D_);
  }
#endif
  int buf = 0;
  for (int kk = 0; kk < D_; kk += KSTEP, buf ^= 1){
#if defined(HAVE_TDM)
    if (wid == 0) wait_tensorcnt0();
    __syncthreads();
    if (wid == 0 && kk + KSTEP < D_){
      tdm_load_2d((unsigned)(uintptr_t)(void*)(Xs + (buf^1)*TILE_E),
                  X + (size_t)gm0*D_ + (kk + KSTEP), D_, (unsigned)mtot, KSTEP, 128, D_);
      tdm_load_2d((unsigned)(uintptr_t)(void*)(Ws + (buf^1)*TILE_E),
                  W + (size_t)gn0*D_ + (kk + KSTEP), D_, D_,             KSTEP, 128, D_);
    }
#else
    { // cooperative fallback copy: 128x32 bf16 tiles, 16B per thread per step
      const int tid = wid*32 + (threadIdx.x & 31);
      for (int e = tid*8; e < TILE_E; e += 256*8){
        const int row = e >> 5, col = e & 31;
        *(v8bf*)&Xs[buf*TILE_E + e] = *(const v8bf*)&X[(size_t)(gm0+row)*D_ + kk + col];
        *(v8bf*)&Ws[buf*TILE_E + e] = *(const v8bf*)&W[(size_t)(gn0+row)*D_ + kk + col];
      }
      __syncthreads();
    }
#endif
    const bf16* xs = Xs + buf*TILE_E;
    const bf16* ws = Ws + buf*TILE_E;
    v16bf af[2];
    #pragma unroll
    for (int mt=0; mt<2; ++mt)
      af[mt] = load_frag_a(&xs[(mw + mt*16 + lm)*KSTEP + lh*16]);
    v16bf bfr[4];
    #pragma unroll
    for (int nt=0; nt<4; ++nt)
      bfr[nt] = load_frag_b(&ws[(nw + nt*16 + lm)*KSTEP + lh*16]);
    #pragma unroll
    for (int mt=0; mt<2; ++mt)
      #pragma unroll
      for (int nt=0; nt<4; ++nt)
        c[mt][nt] = wmma_bf(af[mt], bfr[nt], c[mt][nt]);
    __syncthreads();
  }
}

// ---------------- projection GEMM: out = x @ W^T + bias ----------------
// vmode 0: out[((b*H+h)*S+s)*64+dk] (Q,K)   vmode 2: out[((b*H+h)*64+dk)*S+s] (V transposed)
__global__ __launch_bounds__(256)
void proj_gemm(const bf16* __restrict__ X, const bf16* __restrict__ W,
               const float* __restrict__ bias, bf16* __restrict__ out, int vmode){
  __shared__ alignas(16) bf16 Xs[2][TILE_E];
  __shared__ alignas(16) bf16 Ws[2][TILE_E];
  const int l  = threadIdx.x & 31, lm = l & 15, lh = l >> 4;
  const int wid = threadIdx.x >> 5;
  const int gm0 = (blockIdx.x >> 3) * 128;
  const int gn0 = (blockIdx.x & 7) * 128;
  const int mw = (wid >> 1) * 32;
  const int nw = (wid & 1) * 64;

  v8f c[2][4];
  #pragma unroll
  for (int i0=0;i0<2;++i0)
    #pragma unroll
    for (int j0=0;j0<4;++j0) c[i0][j0] = (v8f){0,0,0,0,0,0,0,0};

  gemm_tiles(X, W, &Xs[0][0], &Ws[0][0], gm0, gn0, mw, nw, lm, lh, wid, c, MROWS);

  #pragma unroll
  for (int mt=0; mt<2; ++mt)
    #pragma unroll
    for (int nt=0; nt<4; ++nt){
      const int gn = gn0 + nw + nt*16 + lm;
      const float bb = bias[gn];
      const int hh = gn >> 6, dk = gn & 63;
      #pragma unroll
      for (int q=0; q<8; ++q){
        const int gm = gm0 + mw + mt*16 + q + 8*lh;
        const int bi = gm >> 11, ss = gm & 2047;
        const float v = c[mt][nt][q] + bb;
        size_t addr;
        if (vmode == 2) addr = ((size_t)(bi*H_ + hh)*DK_ + dk)*(size_t)S_ + ss;
        else            addr = ((size_t)(bi*H_ + hh)*S_  + ss)*(size_t)DK_ + dk;
        out[addr] = (bf16)v;
      }
    }
}

// ---------------- output GEMM: d_out = attn @ Wo^T + bo (fp32 flat) ----------------
__global__ __launch_bounds__(256)
void out_gemm(const bf16* __restrict__ X, const bf16* __restrict__ W,
              const float* __restrict__ bias, float* __restrict__ out){
  __shared__ alignas(16) bf16 Xs[2][TILE_E];
  __shared__ alignas(16) bf16 Ws[2][TILE_E];
  const int l  = threadIdx.x & 31, lm = l & 15, lh = l >> 4;
  const int wid = threadIdx.x >> 5;
  const int gm0 = (blockIdx.x >> 3) * 128;
  const int gn0 = (blockIdx.x & 7) * 128;
  const int mw = (wid >> 1) * 32;
  const int nw = (wid & 1) * 64;

  v8f c[2][4];
  #pragma unroll
  for (int i0=0;i0<2;++i0)
    #pragma unroll
    for (int j0=0;j0<4;++j0) c[i0][j0] = (v8f){0,0,0,0,0,0,0,0};

  gemm_tiles(X, W, &Xs[0][0], &Ws[0][0], gm0, gn0, mw, nw, lm, lh, wid, c, MROWS);

  #pragma unroll
  for (int mt=0; mt<2; ++mt)
    #pragma unroll
    for (int nt=0; nt<4; ++nt){
      const int gn = gn0 + nw + nt*16 + lm;
      const float bb = bias[gn];
      #pragma unroll
      for (int q=0; q<8; ++q){
        const int gm = gm0 + mw + mt*16 + q + 8*lh;
        out[(size_t)gm*D_ + gn] = c[mt][nt][q] + bb;
      }
    }
}

// ---------------- Qrel[bh,i,r] = Q[bh,i,:] . Rk[r,:]  (r = 0..50, stride 64) ----------------
__global__ __launch_bounds__(64)
void qrel_kernel(const bf16* __restrict__ Q, const float* __restrict__ Rk,
                 float* __restrict__ Qrel){
  const int row = blockIdx.x;              // bh*S + i
  __shared__ float qsh[64];
  const int t = threadIdx.x;
  qsh[t] = (float)Q[(size_t)row*DK_ + t];
  __syncthreads();
  if (t < 51){
    float acc = 0.f;
    #pragma unroll
    for (int d=0; d<64; ++d) acc += qsh[d] * Rk[t*64 + d];
    Qrel[(size_t)row*64 + t] = acc;
  }
}

// ---------------- flash attention with relative-position bias + bucket aggregation --------
// grid (BH, S/64), 128 threads (4 waves). Wave owns 16 query rows.
// Pn[bh,i, 0..49] = normalized prob mass in rel-buckets 1..50 (stride 64).
__global__ __launch_bounds__(128)
void flash_attn(const bf16* __restrict__ Q, const bf16* __restrict__ K,
                const bf16* __restrict__ Vt, const float* __restrict__ Qrel,
                float* __restrict__ O, float* __restrict__ Pn){
  const int bh   = blockIdx.x;
  const int wave = threadIdx.x >> 5;
  const int l    = threadIdx.x & 31, lm = l & 15, lh = l >> 4;
  const int i0   = blockIdx.y * 64 + wave * 16;

  const bf16*  Qb = Q  + (size_t)bh * S_ * DK_;
  const bf16*  Kb = K  + (size_t)bh * S_ * DK_;
  const bf16*  Vb = Vt + (size_t)bh * DK_ * S_;
  const float* QR = Qrel + (size_t)bh * S_ * 64;

  __shared__ alignas(16) float pagg[4][16][52];   // buckets 1..50 at idx 0..49
  __shared__ float rowtmp[4][16];
  __shared__ alignas(16) bf16  pbuf[4][16][32];   // P tile (C-layout -> A-layout relayout)

  for (int t = l; t < 16*52; t += 32) pagg[wave][t/52][t%52] = 0.0f;

  const v16bf aq0 = load_frag_a(&Qb[(size_t)(i0 + lm)*DK_      + lh*16]);
  const v16bf aq1 = load_frag_a(&Qb[(size_t)(i0 + lm)*DK_ + 32 + lh*16]);

  float mrun[8], lrun[8];
  #pragma unroll
  for (int q=0; q<8; ++q){ mrun[q] = -1e30f; lrun[q] = 0.0f; }
  v8f o0 = (v8f){0,0,0,0,0,0,0,0}, o1 = o0, o2 = o0, o3 = o0;
  bool pTouched = false;

  const int jend = i0 + 16;                       // exclusive causal bound
  for (int j0 = 0; j0 < jend; j0 += 32){
    #pragma unroll
    for (int sub = 0; sub < 2; ++sub){
      const int j16 = j0 + sub*16;
      if (j16 >= jend){
        #pragma unroll
        for (int q=0; q<8; ++q) pbuf[wave][q + 8*lh][sub*16 + lm] = (bf16)0.0f;
        continue;
      }
      const int jc = j16 + lm;
      v16bf bk0 = load_frag_b(&Kb[(size_t)jc*DK_      + lh*16]);
      v16bf bk1 = load_frag_b(&Kb[(size_t)jc*DK_ + 32 + lh*16]);
      v8f sc = (v8f){0,0,0,0,0,0,0,0};
      sc = wmma_bf(aq0, bk0, sc);
      sc = wmma_bf(aq1, bk1, sc);

      float sv[8];
      #pragma unroll
      for (int q=0; q<8; ++q){
        const int i  = i0 + q + 8*lh;
        const int dj = jc - i;
        if (dj > 0) sv[q] = -1e30f;
        else {
          const int r = (dj <= -50) ? 0 : (dj + 50);
          sv[q] = (sc[q] + QR[(size_t)i*64 + r]) * 0.125f;   // 1/sqrt(64)
        }
      }

      float fs[8]; bool anyresc = false;
      #pragma unroll
      for (int q=0; q<8; ++q){
        float v = sv[q];
        v = fmaxf(v, __shfl_xor(v, 1));
        v = fmaxf(v, __shfl_xor(v, 2));
        v = fmaxf(v, __shfl_xor(v, 4));
        v = fmaxf(v, __shfl_xor(v, 8));
        const float mn = fmaxf(mrun[q], v);
        const float f  = __expf(mrun[q] - mn);
        fs[q] = f; mrun[q] = mn;
        if (f < 1.0f) anyresc = true;
      }
      #pragma unroll
      for (int q=0; q<8; ++q){
        lrun[q] *= fs[q];
        o0[q] *= fs[q]; o1[q] *= fs[q]; o2[q] *= fs[q]; o3[q] *= fs[q];
      }
      if (pTouched && __ballot(anyresc)){
        if (lm == 0){
          #pragma unroll
          for (int q=0; q<8; ++q) rowtmp[wave][q + 8*lh] = fs[q];
        }
        __builtin_amdgcn_wave_barrier();
        for (int t = l; t < 16*50; t += 32)
          pagg[wave][t/50][t%50] *= rowtmp[wave][t/50];
        __builtin_amdgcn_wave_barrier();
      }

      float p[8];
      #pragma unroll
      for (int q=0; q<8; ++q){
        const float e = __expf(sv[q] - mrun[q]);
        float rs = e;
        rs += __shfl_xor(rs, 1); rs += __shfl_xor(rs, 2);
        rs += __shfl_xor(rs, 4); rs += __shfl_xor(rs, 8);
        lrun[q] += rs;
        p[q] = e;
      }

      if (j16 + 15 >= i0 - 49){                 // near-diagonal band: buckets 1..50
        pTouched = true;
        #pragma unroll
        for (int q=0; q<8; ++q){
          const int i  = i0 + q + 8*lh;
          const int dj = jc - i;
          if (dj >= -49 && dj <= 0) pagg[wave][q + 8*lh][dj + 49] += p[q];
        }
      }
      #pragma unroll
      for (int q=0; q<8; ++q) pbuf[wave][q + 8*lh][sub*16 + lm] = (bf16)p[q];
    }

    __builtin_amdgcn_wave_barrier();
    const v16bf ap = load_frag_a(&pbuf[wave][lm][lh*16]);
    v16bf bv0 = load_frag_b(&Vb[(size_t)( 0 + lm)*S_ + j0 + lh*16]);
    v16bf bv1 = load_frag_b(&Vb[(size_t)(16 + lm)*S_ + j0 + lh*16]);
    v16bf bv2 = load_frag_b(&Vb[(size_t)(32 + lm)*S_ + j0 + lh*16]);
    v16bf bv3 = load_frag_b(&Vb[(size_t)(48 + lm)*S_ + j0 + lh*16]);
    o0 = wmma_bf(ap, bv0, o0);
    o1 = wmma_bf(ap, bv1, o1);
    o2 = wmma_bf(ap, bv2, o2);
    o3 = wmma_bf(ap, bv3, o3);
  }

  #pragma unroll
  for (int q=0; q<8; ++q){
    const float inv = 1.0f / lrun[q];
    const size_t rb = ((size_t)bh*S_ + (i0 + q + 8*lh)) * DK_;
    O[rb +  0 + lm] = o0[q] * inv;
    O[rb + 16 + lm] = o1[q] * inv;
    O[rb + 32 + lm] = o2[q] * inv;
    O[rb + 48 + lm] = o3[q] * inv;
  }
  if (lm == 0){
    #pragma unroll
    for (int q=0; q<8; ++q) rowtmp[wave][q + 8*lh] = lrun[q];
  }
  __builtin_amdgcn_wave_barrier();
  for (int t = l; t < 16*50; t += 32){
    const int m = t / 50, r = t % 50;
    Pn[((size_t)bh*S_ + (i0 + m))*64 + r] = pagg[wave][m][r] / rowtmp[wave][m];
  }
}

// ---------------- attn = O + Pagg @ Rv (bucket0 = 1 - sum), head-interleave to [B,S,D] bf16
__global__ __launch_bounds__(64)
void rel_combine(const float* __restrict__ O, const float* __restrict__ Pn,
                 const float* __restrict__ Rv, bf16* __restrict__ attn){
  const int row = blockIdx.x;               // bh*S + i
  const int bh = row >> 11, i = row & (S_-1);
  const int b  = bh >> 4,   h = bh & 15;
  __shared__ float psh[50];
  __shared__ float s_sum;
  const int t = threadIdx.x;
  if (t < 50) psh[t] = Pn[(size_t)row*64 + t];
  __syncthreads();
  if (t == 0){
    float s = 0.f;
    for (int k=0; k<50; ++k) s += psh[k];
    s_sum = s;
  }
  __syncthreads();
  float acc = O[(size_t)row*DK_ + t] + (1.0f - s_sum) * Rv[t];   // bucket 0
  #pragma unroll 10
  for (int k=0; k<50; ++k) acc += psh[k] * Rv[(size_t)(k+1)*DK_ + t];
  attn[((size_t)(b*S_ + i))*D_ + h*DK_ + t] = (bf16)acc;
}

// ---------------- host launch ----------------
extern "C" void kernel_launch(void* const* d_in, const int* in_sizes, int n_in,
                              void* d_out, int out_size, void* d_ws, size_t ws_size,
                              hipStream_t stream){
  (void)in_sizes; (void)n_in; (void)out_size; (void)ws_size;
  const float* x  = (const float*)d_in[0];
  // d_in[1] = causal_mask (causality computed analytically)
  const float* Wq = (const float*)d_in[2];
  const float* bq = (const float*)d_in[3];
  const float* Wk = (const float*)d_in[4];
  const float* bk = (const float*)d_in[5];
  const float* Wv = (const float*)d_in[6];
  const float* bv = (const float*)d_in[7];
  const float* Wo = (const float*)d_in[8];
  const float* bo = (const float*)d_in[9];
  const float* Rk = (const float*)d_in[10];
  const float* Rv = (const float*)d_in[11];
  float* out = (float*)d_out;

  char* ws = (char*)d_ws;
  size_t off = 0;
  auto carve = [&](size_t bytes) -> char* {
    char* p = ws + off;
    off = (off + bytes + 255) & ~(size_t)255;
    return p;
  };
  bf16*  xbf   = (bf16*) carve((size_t)MROWS*D_*2);
  bf16*  wqbf  = (bf16*) carve((size_t)D_*D_*2);
  bf16*  wkbf  = (bf16*) carve((size_t)D_*D_*2);
  bf16*  wvbf  = (bf16*) carve((size_t)D_*D_*2);
  bf16*  wobf  = (bf16*) carve((size_t)D_*D_*2);
  bf16*  qbf   = (bf16*) carve((size_t)BH_*S_*DK_*2);
  bf16*  kbf   = (bf16*) carve((size_t)BH_*S_*DK_*2);
  bf16*  vtbf  = (bf16*) carve((size_t)BH_*DK_*S_*2);
  float* qrel  = (float*)carve((size_t)BH_*S_*64*4);
  float* obuf  = (float*)carve((size_t)BH_*S_*DK_*4);
  float* pn    = (float*)carve((size_t)BH_*S_*64*4);
  bf16*  atnbf = (bf16*) carve((size_t)MROWS*D_*2);

  int n = MROWS * D_;
  cvt_bf16<<<(n+255)/256, 256, 0, stream>>>(x,  xbf,  n);
  n = D_ * D_;
  cvt_bf16<<<(n+255)/256, 256, 0, stream>>>(Wq, wqbf, n);
  cvt_bf16<<<(n+255)/256, 256, 0, stream>>>(Wk, wkbf, n);
  cvt_bf16<<<(n+255)/256, 256, 0, stream>>>(Wv, wvbf, n);
  cvt_bf16<<<(n+255)/256, 256, 0, stream>>>(Wo, wobf, n);

  proj_gemm<<<256, 256, 0, stream>>>(xbf, wqbf, bq, qbf,  0);
  proj_gemm<<<256, 256, 0, stream>>>(xbf, wkbf, bk, kbf,  0);
  proj_gemm<<<256, 256, 0, stream>>>(xbf, wvbf, bv, vtbf, 2);

  qrel_kernel<<<BH_*S_, 64, 0, stream>>>(qbf, Rk, qrel);

  flash_attn<<<dim3(BH_, S_/64), 128, 0, stream>>>(qbf, kbf, vtbf, qrel, obuf, pn);

  rel_combine<<<BH_*S_, 64, 0, stream>>>(obuf, pn, Rv, atnbf);

  out_gemm<<<256, 256, 0, stream>>>(atnbf, wobf, bo, out);
}
